// SingleLayer_47107201303280
// MI455X (gfx1250) — matrix operand order repaired and checked by reading, not compiled, not run on hardware
//
#include <hip/hip_runtime.h>
#include <hip/hip_bf16.h>

#define NB   16384   // batch
#define NC   72      // channels in x
#define NCP  96      // channel dim padded for K-reorder (zeros in 72..95)
#define NH   34      // height
#define NP   36      // padded height positions (p = h+1; p=0,35 are zero)
#define OUTC 12      // conv output channels
#define NCO  84      // output channels (72 + 12)

typedef __attribute__((ext_vector_type(16))) __bf16 v16bf;
typedef __attribute__((ext_vector_type(8)))  __bf16 v8bf;
typedef __attribute__((ext_vector_type(8)))  float  v8f;

// ---------------------------------------------------------------------------
// Kernel 0: zero BN accumulators
// ---------------------------------------------------------------------------
__global__ void zero_kernel(float* __restrict__ p, int n) {
    int i = blockIdx.x * blockDim.x + threadIdx.x;
    if (i < n) p[i] = 0.0f;
}

// ---------------------------------------------------------------------------
// Kernel 1: per-sample gates + global BN stats (sum / sumsq per xxx-channel)
// 288 threads = 4 samples x 72 channels
// ---------------------------------------------------------------------------
__global__ __launch_bounds__(288)
void gates_stats_kernel(const float* __restrict__ x,
                        const float* __restrict__ fc_w,
                        const float* __restrict__ fc_bias,
                        float* __restrict__ ws_gates,   // [NB][5]
                        float* __restrict__ gsum,       // [NC]
                        float* __restrict__ gssq)       // [NC]
{
    __shared__ float wlds[NH];
    __shared__ float sw_l[4 * NC];
    __shared__ float dg[4 * 6];
    __shared__ float gl[4 * 5];
    __shared__ float lsum[NC];
    __shared__ float lssq[NC];

    const int t = threadIdx.x;
    const int s = t / NC;            // 0..3 sample-in-block
    const int c = t % NC;            // 0..71 channel
    const int b = blockIdx.x * 4 + s;

    if (t < NH) wlds[t] = fc_w[t];
    if (t < NC) { lsum[t] = 0.0f; lssq[t] = 0.0f; }
    __syncthreads();

    const float* xp = x + ((size_t)b * NC + c) * NH;
    float s1 = 0.0f, s2 = 0.0f, sw = 0.0f;
    #pragma unroll
    for (int h = 0; h < NH; ++h) {
        float v = xp[h];
        s1 += v;
        s2 += v * v;
        sw += v * wlds[h];
    }
    sw_l[s * NC + c] = sw;
    __syncthreads();

    if (c < 6) {
        float acc = 0.0f;
        #pragma unroll
        for (int cc = 0; cc < 12; ++cc) acc += sw_l[s * NC + c * 12 + cc];
        dg[s * 6 + c] = acc;
    }
    __syncthreads();

    if (c == 0) {
        float bias    = fc_bias[0];
        float lin_out = dg[s * 6 + 5] * (1.0f / 12.0f) + bias;  // group 4 (last)
        float lin_x11 = dg[s * 6 + 0] * (1.0f / 12.0f) + bias;  // x1 group
        float g1 = 1.0f / (1.0f + __expf(-(lin_out + lin_x11)));
        gl[s * 5 + 0] = g1;
        #pragma unroll
        for (int j = 0; j < 4; ++j) {
            float lm = dg[s * 6 + 1 + j] * (1.0f / 12.0f) + bias;
            gl[s * 5 + 1 + j] = 1.0f / (1.0f + __expf(-(lm + lin_out)));
        }
        #pragma unroll
        for (int j = 0; j < 5; ++j) ws_gates[(size_t)b * 5 + j] = gl[s * 5 + j];
    }
    __syncthreads();

    // source channel -> dest xxx-channel + gate factor; accumulate stats
    float f; int cp;
    if (c < 12)       { f = gl[s * 5 + 0]; cp = c; }
    else if (c < 60)  { int gj = (c - 12) / 12;
                        f = gl[s * 5 + 1 + gj];
                        cp = 12 + (3 - gj) * 12 + (c - 12) % 12; }
    else              { f = 1.0f; cp = c; }

    atomicAdd(&lsum[cp], f * s1);
    atomicAdd(&lssq[cp], f * f * s2);
    __syncthreads();

    if (t < NC) {
        atomicAdd(&gsum[t], lsum[t]);
        atomicAdd(&gssq[t], lssq[t]);
    }
}

// ---------------------------------------------------------------------------
// Kernel 2: fold BN into per-channel scale/shift
// ---------------------------------------------------------------------------
__global__ void finalize_kernel(const float* __restrict__ gsum,
                                const float* __restrict__ gssq,
                                const float* __restrict__ gamma,
                                const float* __restrict__ beta,
                                float* __restrict__ scale,
                                float* __restrict__ shift)
{
    int c = threadIdx.x;
    if (c < NC) {
        const float invN = 1.0f / (float)((long long)NB * NH);
        float mu  = gsum[c] * invN;
        float var = gssq[c] * invN - mu * mu;
        float sc  = gamma[c] * rsqrtf(var + 1e-5f);
        scale[c] = sc;
        shift[c] = beta[c] - mu * sc;
    }
}

// ---------------------------------------------------------------------------
// Kernel 3: gating + BN + ReLU staged position-major in LDS (bf16), then
// WMMA conv with tap-major K (K = tap*96 + c, 9 k-steps of 32) so each
// lane's A-fragment is two contiguous 16B LDS loads per k-step.
// All 18 A loads per tile are hoisted ahead of the WMMA chain, and the
// accumulator is split into 3 independent chains to avoid serialization.
// Block = 256 threads (8 waves), 8 samples. Tile M=16 = 8 samples x 2 h.
// ---------------------------------------------------------------------------
__global__ __launch_bounds__(256)
void bn_conv_kernel(const float* __restrict__ x,
                    const float* __restrict__ conv_w,    // [12][72][3]
                    const float* __restrict__ ws_gates,  // [NB][5]
                    const float* __restrict__ scale,
                    const float* __restrict__ shift,
                    float* __restrict__ out)             // [NB][84][34]
{
    // position-major staging: hbuf[s][p][c], p=h+1 (p=0,35 zero), c padded 96
    __shared__ __attribute__((aligned(16))) __bf16 hbuf[8][NP][NCP]; // 54 KB
    __shared__ float g_l[8][5];
    __shared__ float sc_l[NC], sh_l[NC];

    const int t  = threadIdx.x;
    const int b0 = blockIdx.x * 8;

    if (t < NC) { sc_l[t] = scale[t]; sh_l[t] = shift[t]; }
    if (t < 40) g_l[t / 5][t % 5] = ws_gates[(size_t)(b0 + t / 5) * 5 + t % 5];

    // zero entire staging buffer (covers h-pad rows and channel pad 72..95)
    {
        uint4* z = (uint4*)&hbuf[0][0][0];
        const int NZ = (8 * NP * NCP * 2) / 16;   // 3456 uint4
        for (int i = t; i < NZ; i += 256) z[i] = make_uint4(0u, 0u, 0u, 0u);
    }
    __syncthreads();

    const int SAMP = NC * NH;  // 2448

    // single pass over x: passthrough out[:, :72] + gated/BN/ReLU -> LDS
    for (int idx = t; idx < 8 * SAMP; idx += 256) {
        int s   = idx / SAMP;
        int rem = idx - s * SAMP;
        int src = rem / NH;          // source channel
        int h   = rem - src * NH;
        float v = x[(size_t)(b0 + s) * SAMP + rem];
        out[(size_t)(b0 + s) * (NCO * NH) + rem] = v;   // passthrough

        int cp; float f;
        if (src < 12)      { cp = src; f = g_l[s][0]; }
        else if (src < 60) { int gj = (src - 12) / 12;
                             cp = 12 + (3 - gj) * 12 + (src - 12) % 12;
                             f = g_l[s][1 + gj]; }
        else               { cp = src; f = 1.0f; }
        float hn = fmaxf(sc_l[cp] * (v * f) + sh_l[cp], 0.0f);
        hbuf[s][h + 1][cp] = (__bf16)hn;
    }
    __syncthreads();

    const int wave  = t >> 5;
    const int lane  = t & 31;
    const int n     = lane & 15;   // B column = output channel
    const int khalf = lane >> 4;

    // B fragments (weights), K = tap*96 + c : ks = tap*3 + cc
    v16bf bfrag[9];
    #pragma unroll
    for (int ks = 0; ks < 9; ++ks) {
        const int tap = ks / 3, cc = ks % 3;
        v16bf bb;
        #pragma unroll
        for (int e = 0; e < 16; ++e) {
            int c = cc * 32 + khalf * 8 + (e & 7) + (e >> 3) * 16;   // B element e -> Klocal khalf*8+(e&7)+(e>>3)*16
            float wv = 0.0f;
            if (n < OUTC && c < NC) wv = conv_w[(n * NC + c) * 3 + tap];
            bb[e] = (__bf16)wv;
        }
        bfrag[ks] = bb;
    }

    // A row m: sample = m>>1, h-parity = m&1
    const int m    = lane & 15;
    const int samp = m >> 1;
    const int par  = m & 1;

    for (int tile = wave; tile < 17; tile += 8) {
        const int h0 = tile * 2;

        // hoist all 18 16B LDS loads for this tile ahead of the WMMA chain
        v8bf lo[9], hi[9];
        #pragma unroll
        for (int ks = 0; ks < 9; ++ks) {
            const int tap = ks / 3, cc = ks % 3;
            const v8bf* row = (const v8bf*)&hbuf[samp][h0 + par + tap][0];
            lo[ks] = row[cc * 4 + khalf];        // channels cc*32 + khalf*8 + 0..7
            hi[ks] = row[cc * 4 + khalf + 2];    // +16 channels
        }

        // 3 independent accumulator chains, 3 WMMAs each
        v8f a0 = {}, a1 = {}, a2 = {};
        #pragma unroll
        for (int g = 0; g < 3; ++g) {
            const int k0 = g * 3;
            v16bf aa0 = __builtin_shufflevector(lo[k0 + 0], hi[k0 + 0],
                        0, 1, 2, 3, 4, 5, 6, 7, 8, 9, 10, 11, 12, 13, 14, 15);
            v16bf aa1 = __builtin_shufflevector(lo[k0 + 1], hi[k0 + 1],
                        0, 1, 2, 3, 4, 5, 6, 7, 8, 9, 10, 11, 12, 13, 14, 15);
            v16bf aa2 = __builtin_shufflevector(lo[k0 + 2], hi[k0 + 2],
                        0, 1, 2, 3, 4, 5, 6, 7, 8, 9, 10, 11, 12, 13, 14, 15);
            a0 = __builtin_amdgcn_wmma_f32_16x16x32_bf16(
                     false, aa0, false, bfrag[k0 + 0], (short)0, a0, false, false);
            a1 = __builtin_amdgcn_wmma_f32_16x16x32_bf16(
                     false, aa1, false, bfrag[k0 + 1], (short)0, a1, false, false);
            a2 = __builtin_amdgcn_wmma_f32_16x16x32_bf16(
                     false, aa2, false, bfrag[k0 + 2], (short)0, a2, false, false);
        }
        v8f acc = (a0 + a1) + a2;

        if (n < OUTC) {
            #pragma unroll
            for (int i = 0; i < 8; ++i) {
                int mr = i + khalf * 8;            // D row
                int ss = mr >> 1, pp = mr & 1;
                int hh = h0 + pp;                  // always < 34 (17 tiles x 2)
                out[((size_t)(b0 + ss) * NCO + NC + n) * NH + hh] = acc[i];
            }
        }
    }
}

// ---------------------------------------------------------------------------
extern "C" void kernel_launch(void* const* d_in, const int* in_sizes, int n_in,
                              void* d_out, int out_size, void* d_ws, size_t ws_size,
                              hipStream_t stream)
{
    const float* x       = (const float*)d_in[0];
    const float* gamma   = (const float*)d_in[1];
    const float* beta    = (const float*)d_in[2];
    const float* conv_w  = (const float*)d_in[3];
    const float* fc_w    = (const float*)d_in[4];
    const float* fc_bias = (const float*)d_in[5];
    float* out = (float*)d_out;
    float* ws  = (float*)d_ws;

    float* ws_gates = ws;                 // NB*5 floats
    float* gsum     = ws + (size_t)NB * 5;
    float* gssq     = gsum + NC;
    float* scl      = gssq + NC;
    float* shf      = scl + NC;

    zero_kernel<<<1, 256, 0, stream>>>(gsum, 2 * NC);
    gates_stats_kernel<<<NB / 4, 288, 0, stream>>>(x, fc_w, fc_bias,
                                                   ws_gates, gsum, gssq);
    finalize_kernel<<<1, 128, 0, stream>>>(gsum, gssq, gamma, beta, scl, shf);
    bn_conv_kernel<<<NB / 8, 256, 0, stream>>>(x, conv_w, ws_gates, scl, shf, out);
}
